// LSTM_62277025792381
// MI455X (gfx1250) — matrix operand order repaired and checked by reading, not compiled, not run on hardware
//
#include <hip/hip_runtime.h>
#include <stdint.h>

// Problem constants (from reference): T=512, B=64, D=256, H=256, 4H=1024
#define TT 512
#define BB 64
#define DI 256
#define HH 256
#define G4 1024

// Workspace layout (bytes):
//   gxP  fp32, WMMA C-fragment packed  : 134,217,728
//   Xbf  bf16 [T*B, 256]               :  16,777,216
//   WxP  packed bf16 B-fragments       :     524,288
//   WhP  packed bf16 B-fragments       :     524,288

#if defined(__AMDGCN__) && __has_builtin(__builtin_amdgcn_global_load_async_to_lds_b128)
#define HAVE_ASYNC 1
#else
#define HAVE_ASYNC 0
#endif

typedef __attribute__((ext_vector_type(16))) __bf16 v16bf;
typedef __attribute__((ext_vector_type(8)))  float  v8f;
typedef __attribute__((ext_vector_type(4)))  int    v4i;

union Frag { v16bf v; uint4 q[2]; };
union Acc8 { v8f  v; uint4 q[2]; };

__device__ __forceinline__ unsigned short f2bf(float f) {
  unsigned int u = __float_as_uint(f);
  u += 0x7FFFu + ((u >> 16) & 1u);          // round-to-nearest-even
  return (unsigned short)(u >> 16);
}

__device__ __forceinline__ float sigm(float x) {
  return 1.0f / (1.0f + __expf(-x));
}

#if HAVE_ASYNC
typedef __attribute__((address_space(1))) v4i gv4i;   // global, 128-bit payload
typedef __attribute__((address_space(3))) v4i lv4i;   // LDS, 128-bit payload

__device__ __forceinline__ void async_cp16(const void* g, void* l) {
  __builtin_amdgcn_global_load_async_to_lds_b128((gv4i*)(void*)g, (lv4i*)l, 0, 0);
}
__device__ __forceinline__ void wait_async_16() {
#if __has_builtin(__builtin_amdgcn_s_wait_asynccnt)
  __builtin_amdgcn_s_wait_asynccnt(16);
#else
  asm volatile("s_wait_asynccnt 0x10" ::: "memory");
#endif
}
__device__ __forceinline__ void wait_async_0() {
#if __has_builtin(__builtin_amdgcn_s_wait_asynccnt)
  __builtin_amdgcn_s_wait_asynccnt(0);
#else
  asm volatile("s_wait_asynccnt 0x0" ::: "memory");
#endif
}
#endif

// ---------------------------------------------------------------------------
// Pack a [256, 4*256] weight (four [256,256] gate matrices) into the per-lane
// WMMA B-operand layout for v_wmma_f32_16x16x32_bf16:
//   fragment (ntile, ktile): lane -> column N = ntile*16 + (lane&15),
//   element e (0..15)       -> K = ktile*32 + (e&7) + 8*(lane>=16) + 16*(e>=8).
// Each lane's 16 bf16 (32 B) contiguous -> two b128 loads.
// ---------------------------------------------------------------------------
__global__ void pack_w_kernel(const float* __restrict__ W0, const float* __restrict__ W1,
                              const float* __restrict__ W2, const float* __restrict__ W3,
                              unsigned short* __restrict__ dst) {
  int tid  = blockIdx.x * blockDim.x + threadIdx.x;   // 0 .. 64*8*32-1
  int lane = tid & 31;
  int kt   = (tid >> 5) & 7;
  int nt   = tid >> 8;
  int n    = nt * 16 + (lane & 15);
  int gate = n >> 8;
  int ncol = n & 255;
  const float* W = (gate == 0) ? W0 : (gate == 1) ? W1 : (gate == 2) ? W2 : W3;
  int hi = lane >> 4;
  unsigned short* out = dst + (size_t)tid * 16;
#pragma unroll
  for (int e = 0; e < 16; ++e) {
    int k = kt * 32 + (e & 7) + hi * 8 + (e >> 3) * 16;
    out[e] = f2bf(W[(size_t)k * HH + ncol]);
  }
}

__global__ void xconv_kernel(const float* __restrict__ X, unsigned short* __restrict__ Xb, int n) {
  int i = blockIdx.x * blockDim.x + threadIdx.x;
  if (i < n) Xb[i] = f2bf(X[i]);
}

// ---------------------------------------------------------------------------
// Phase 1: gx = X @ Wx + bias. Output stored PACKED in the C-fragment per-lane
// layout: tile id = mtile*64 + nt, lane's 8 floats contiguous:
//   gxP[ (tileid*32 + lane)*8 + v ]  with  M = v + 8*(lane>=16), N = lane&15.
// Makes both the store here and the scan's accumulator seed pure b128 traffic;
// one (t,blk) slab of gx is a contiguous 64 KB block.
// ---------------------------------------------------------------------------
__global__ __launch_bounds__(256) void gemm_gx_kernel(
    const unsigned short* __restrict__ Xb, const unsigned short* __restrict__ WxP,
    const float* __restrict__ b0, const float* __restrict__ b1,
    const float* __restrict__ b2, const float* __restrict__ b3,
    float* __restrict__ gxP) {
  int lane  = threadIdx.x & 31;
  int wid   = blockIdx.x * 8 + (threadIdx.x >> 5);  // 0..32767
  int mtile = wid >> 4;                             // 0..2047
  int ngrp  = wid & 15;                             // ntiles ngrp*4 .. +3
  int hi    = lane >> 4;
  int m     = mtile * 16 + (lane & 15);

  v8f acc[4];
#pragma unroll
  for (int j = 0; j < 4; ++j) acc[j] = (v8f)0.0f;

  for (int kt = 0; kt < 8; ++kt) {
    Frag a;
    const uint4* ap = (const uint4*)(Xb + (size_t)m * DI + kt * 32 + hi * 8);
    a.q[0] = ap[0];
    a.q[1] = ap[2];
#pragma unroll
    for (int j = 0; j < 4; ++j) {
      int nt = ngrp * 4 + j;
      Frag b;
      const uint4* bp = (const uint4*)(WxP + (((size_t)nt * 8 + kt) * 32 + lane) * 16);
      b.q[0] = bp[0];
      b.q[1] = bp[1];
      acc[j] = __builtin_amdgcn_wmma_f32_16x16x32_bf16(
          false, a.v, false, b.v, (short)0, acc[j], false, false);
    }
  }

#pragma unroll
  for (int j = 0; j < 4; ++j) {
    int nt   = ngrp * 4 + j;
    int n    = nt * 16 + (lane & 15);
    int gate = n >> 8;
    int ncol = n & 255;
    const float* bias = (gate == 0) ? b0 : (gate == 1) ? b1 : (gate == 2) ? b2 : b3;
    float bv = bias[ncol];
    Acc8 r;
#pragma unroll
    for (int v = 0; v < 8; ++v) r.v[v] = acc[j][v] + bv;
    uint4* op = (uint4*)(gxP + ((size_t)(mtile * 64 + nt) * 32 + lane) * 8);
    op[0] = r.q[0];
    op[1] = r.q[1];
  }
}

// ---------------------------------------------------------------------------
// Phase 2: sequential scan. 4 WGs x 16 batch rows; 8 waves; wave w owns h-col
// tiles {w, w+8} and the matching a/i/f/o gate tiles (gate math in-register).
// - Wh B-fragments: software-pipelined double buffer over the K loop.
// - gx(t+1): async-copied into a 2x64KB LDS double buffer (each wave copies
//   only its own 8 KB slice -> no barrier, waits via s_wait_asynccnt).
// ---------------------------------------------------------------------------
__global__ __launch_bounds__(256) void lstm_scan_kernel(
    const float* __restrict__ gxP, const unsigned short* __restrict__ WhP,
    float* __restrict__ out) {
  __shared__ __align__(16) unsigned short hbuf[16 * 264];
#if HAVE_ASYNC
  __shared__ __align__(16) char gstage[2][8][8192];
#endif

  int lane = threadIdx.x & 31;
  int w    = threadIdx.x >> 5;
  int hi   = lane >> 4;
  int mrow = lane & 15;
  int blk  = blockIdx.x;

  for (int i = threadIdx.x; i < 16 * 264; i += 256) hbuf[i] = 0;
  __syncthreads();

  float s[2][8];
#pragma unroll
  for (int j = 0; j < 2; ++j)
#pragma unroll
    for (int v = 0; v < 8; ++v) s[j][v] = 0.0f;

  int ntb[2] = { w, w + 8 };
  int nts[8];                       // jj = j*4+g  ->  nt = g*16 + ntb[j]
#pragma unroll
  for (int jj = 0; jj < 8; ++jj) nts[jj] = (jj & 3) * 16 + ntb[jj >> 2];

#if HAVE_ASYNC
  // Kick off the copy of gx(t=0) for this wave's tiles.
  {
    int mt0 = blk;                  // mtile(t=0) = 0*4 + blk
#pragma unroll
    for (int jj = 0; jj < 8; ++jj) {
      const float* src = gxP + ((size_t)(mt0 * 64 + nts[jj]) * 32 + lane) * 8;
      char* dst = &gstage[0][w][jj * 1024 + lane * 32];
      async_cp16(src, dst);
      async_cp16(src + 4, dst + 16);
    }
  }
#endif

  for (int t = 0; t < TT; ++t) {
    int mtile = t * 4 + blk;

    v8f acc[2][4];
#if HAVE_ASYNC
    if (t + 1 < TT) {               // prefetch gx(t+1) into the other buffer
      int mt1 = mtile + 4;
#pragma unroll
      for (int jj = 0; jj < 8; ++jj) {
        const float* src = gxP + ((size_t)(mt1 * 64 + nts[jj]) * 32 + lane) * 8;
        char* dst = &gstage[(t + 1) & 1][w][jj * 1024 + lane * 32];
        async_cp16(src, dst);
        async_cp16(src + 4, dst + 16);
      }
      wait_async_16();              // step t's 16 copies are done; t+1 in flight
    } else {
      wait_async_0();
    }
#pragma unroll
    for (int jj = 0; jj < 8; ++jj) {
      const uint4* p = (const uint4*)&gstage[t & 1][w][jj * 1024 + lane * 32];
      Acc8 r;
      r.q[0] = p[0];
      r.q[1] = p[1];
      acc[jj >> 2][jj & 3] = r.v;
    }
#else
#pragma unroll
    for (int jj = 0; jj < 8; ++jj) {
      const uint4* p = (const uint4*)(gxP + ((size_t)(mtile * 64 + nts[jj]) * 32 + lane) * 8);
      Acc8 r;
      r.q[0] = p[0];
      r.q[1] = p[1];
      acc[jj >> 2][jj & 3] = r.v;
    }
    if (t + 1 < TT) {               // warm caches for next step's seeds
#pragma unroll
      for (int jj = 0; jj < 8; ++jj)
        __builtin_prefetch(gxP + ((size_t)((mtile + 4) * 64 + nts[jj]) * 32 + lane) * 8, 0, 0);
    }
#endif

    // g += h_prev @ Wh : K loop of 8, double-buffered B fragments
    Frag bcur[8], bnext[8], acur, anext;
    {
      const uint4* ap = (const uint4*)(&hbuf[mrow * 264 + hi * 8]);
      acur.q[0] = ap[0];
      acur.q[1] = ap[2];
#pragma unroll
      for (int jj = 0; jj < 8; ++jj) {
        const uint4* bp = (const uint4*)(WhP + ((size_t)nts[jj] * 8 * 32 + lane) * 16);
        bcur[jj].q[0] = bp[0];
        bcur[jj].q[1] = bp[1];
      }
    }
#pragma unroll
    for (int kt = 0; kt < 8; ++kt) {
      if (kt < 7) {
        const uint4* ap = (const uint4*)(&hbuf[mrow * 264 + (kt + 1) * 32 + hi * 8]);
        anext.q[0] = ap[0];
        anext.q[1] = ap[2];
#pragma unroll
        for (int jj = 0; jj < 8; ++jj) {
          const uint4* bp =
              (const uint4*)(WhP + (((size_t)nts[jj] * 8 + (kt + 1)) * 32 + lane) * 16);
          bnext[jj].q[0] = bp[0];
          bnext[jj].q[1] = bp[1];
        }
      }
#pragma unroll
      for (int jj = 0; jj < 8; ++jj)
        acc[jj >> 2][jj & 3] = __builtin_amdgcn_wmma_f32_16x16x32_bf16(
            false, acur.v, false, bcur[jj].v, (short)0, acc[jj >> 2][jj & 3], false, false);
      if (kt < 7) {
        acur = anext;
#pragma unroll
        for (int jj = 0; jj < 8; ++jj) bcur[jj] = bnext[jj];
      }
    }

    __syncthreads();   // everyone done reading h_prev from LDS

#pragma unroll
    for (int j = 0; j < 2; ++j) {
      int hcol = ntb[j] * 16 + mrow;
#pragma unroll
      for (int v = 0; v < 8; ++v) {
        float av = tanhf(acc[j][0][v]);
        float iv = sigm(acc[j][1][v]);
        float fv = sigm(acc[j][2][v]);
        float ov = sigm(acc[j][3][v]);
        float sn = av * iv + s[j][v] * fv;
        s[j][v]  = sn;
        float h  = tanhf(sn) * ov;
        int M    = v + hi * 8;
        hbuf[M * 264 + hcol] = f2bf(h);
        out[((size_t)t * BB + blk * 16 + M) * HH + hcol] = h;
      }
    }

    __syncthreads();   // h ready for next timestep
  }
}

// ---------------------------------------------------------------------------
extern "C" void kernel_launch(void* const* d_in, const int* in_sizes, int n_in,
                              void* d_out, int out_size, void* d_ws, size_t ws_size,
                              hipStream_t stream) {
  (void)in_sizes; (void)n_in; (void)out_size; (void)ws_size;

  const float* X    = (const float*)d_in[0];
  const float* Wa_x = (const float*)d_in[1];
  const float* Wi_x = (const float*)d_in[2];
  const float* Wf_x = (const float*)d_in[3];
  const float* Wo_x = (const float*)d_in[4];
  const float* Wa_h = (const float*)d_in[5];
  const float* Wi_h = (const float*)d_in[6];
  const float* Wf_h = (const float*)d_in[7];
  const float* Wo_h = (const float*)d_in[8];
  const float* ba   = (const float*)d_in[9];
  const float* bi   = (const float*)d_in[10];
  const float* bf   = (const float*)d_in[11];
  const float* bo   = (const float*)d_in[12];

  char* ws = (char*)d_ws;
  float*          gxP = (float*)ws;                                  // 134,217,728 B
  unsigned short* Xb  = (unsigned short*)(ws + (size_t)TT * BB * G4 * 4);
  unsigned short* WxP = Xb + (size_t)TT * BB * DI;                   // +16,777,216 B
  unsigned short* WhP = WxP + (size_t)64 * 8 * 32 * 16;              // +524,288 B

  pack_w_kernel<<<64, 256, 0, stream>>>(Wa_x, Wi_x, Wf_x, Wo_x, WxP);
  pack_w_kernel<<<64, 256, 0, stream>>>(Wa_h, Wi_h, Wf_h, Wo_h, WhP);

  int nX = TT * BB * DI;
  xconv_kernel<<<(nX + 255) / 256, 256, 0, stream>>>(X, Xb, nX);

  gemm_gx_kernel<<<4096, 256, 0, stream>>>(Xb, WxP, ba, bi, bf, bo, gxP);

  lstm_scan_kernel<<<4, 256, 0, stream>>>(gxP, WhP, (float*)d_out);
}
